// QuantLSTMLayer_18975165514460
// MI455X (gfx1250) — matrix operand order repaired and checked by reading, not compiled, not run on hardware
//
#include <hip/hip_runtime.h>
#include <hip/hip_bf16.h>
#include <stdint.h>

// ---------------------------------------------------------------------------
// QuantLSTM (identity quant): for t in 0..511:
//   gates = x_t @ Wi + h @ Wh + b ; i,f,o=sigmoid, a=tanh
//   c = f*c + i*a ; h = o*tanh(c)
// MI455X strategy:
//   K0: pack 8 weight mats f32->bf16 (concat [4H][K], k-major)
//   K1: pack inputs f32->bf16
//   K2: init h(bf16) state + grid-barrier counter
//   K3: bulk GEMM X = in @ Wi + b (bf16 WMMA, async-to-LDS double buffer)
//   K4: ONE persistent kernel runs all 512 recurrent steps:
//       - Wh slice resident in LDS (128KB) for the whole sequence
//       - h streamed to LDS each step via global_load_async_to_lds_b128,
//         pipelined with s_wait_asynccnt 8
//       - c state in registers across all steps
//       - inter-WG sync via monotonic L2 atomic + s_sleep
// ---------------------------------------------------------------------------

typedef __attribute__((ext_vector_type(16))) __bf16 v16bf;
typedef __attribute__((ext_vector_type(8)))  __bf16 v8bf;
typedef __attribute__((ext_vector_type(4)))  __bf16 v4bf;
typedef __attribute__((ext_vector_type(8)))  float  v8f;

#define T_STEPS 512
#define BATCH   64
#define HID     1024
#define FOURH   4096
#define KDIM    1024
#define MTOT    (T_STEPS * BATCH)   // 32768
#define NWG     64                  // persistent workgroups

static __device__ __forceinline__ v16bf concat8(v8bf lo, v8bf hi) {
  return __builtin_shufflevector(lo, hi, 0,1,2,3,4,5,6,7,8,9,10,11,12,13,14,15);
}
static __device__ __forceinline__ float sigmoidf_(float x) {
  return 1.0f / (1.0f + __expf(-x));
}

// CDNA5 async copy: global -> LDS, 16B per lane, tracked by ASYNCcnt.
static __device__ __forceinline__ void async_load_b128(const void* gptr,
                                                       void* lptr) {
  unsigned loff = (unsigned)(uintptr_t)
      (__attribute__((address_space(3))) void*)lptr;   // LDS byte address
  asm volatile("global_load_async_to_lds_b128 %0, %1, off"
               :: "v"(loff), "v"(gptr)
               : "memory");
}
static __device__ __forceinline__ void wait_async0() {
  asm volatile("s_wait_asynccnt 0" ::: "memory");
}
static __device__ __forceinline__ void wait_async8() {  // allow 8 in flight
  asm volatile("s_wait_asynccnt 8" ::: "memory");
}

// --------------------------- K0: weight pack (f32 -> bf16) ------------------
__global__ __launch_bounds__(256) void pack_weights_kernel(
    const float* __restrict__ w0, const float* __restrict__ w1,
    const float* __restrict__ w2, const float* __restrict__ w3,
    const float* __restrict__ w4, const float* __restrict__ w5,
    const float* __restrict__ w6, const float* __restrict__ w7,
    __bf16* __restrict__ Wi, __bf16* __restrict__ Wh) {
  size_t idx = ((size_t)blockIdx.x * 256 + threadIdx.x) * 4;  // 8M elems
  int    sec = (int)(idx >> 20);
  size_t off = idx & ((1u << 20) - 1);
  const float* srcs[8] = {w0, w1, w2, w3, w4, w5, w6, w7};
  const float4 v = *(const float4*)(srcs[sec] + off);
  v4bf o = { (__bf16)v.x, (__bf16)v.y, (__bf16)v.z, (__bf16)v.w };
  __bf16* dst = (sec < 4) ? (Wi + idx) : (Wh + (idx - (size_t)4 * 1048576));
  *(v4bf*)dst = o;
}

// --------------------------- K1: input pack (f32 -> bf16) -------------------
__global__ __launch_bounds__(256) void pack_inputs_kernel(
    const float* __restrict__ in, __bf16* __restrict__ out) {
  size_t idx = ((size_t)blockIdx.x * 256 + threadIdx.x) * 4;
  const float4 v = *(const float4*)(in + idx);
  v4bf o = { (__bf16)v.x, (__bf16)v.y, (__bf16)v.z, (__bf16)v.w };
  *(v4bf*)(out + idx) = o;
}

// --------------------------- K2: state init --------------------------------
__global__ __launch_bounds__(256) void init_state_kernel(
    const float* __restrict__ hx, __bf16* __restrict__ h0,
    unsigned* __restrict__ bar) {
  int i = blockIdx.x * 256 + threadIdx.x;  // 65536 elements
  h0[i] = (__bf16)hx[i];
  if (i == 0) *bar = 0u;
}

// --------------------------- K3: bulk input GEMM ---------------------------
// X[MTOT][4H] = Ab @ Wi^T + bias.  256 thr = 8 waves, tile 128x128,
// K-chunks of 32, async-to-LDS double buffer.
__global__ __launch_bounds__(256) void gemm_x_kernel(
    const __bf16* __restrict__ Ab,   // [32768][1024] bf16
    const __bf16* __restrict__ Bw,   // [4096][1024] bf16 (n-major)
    const float* __restrict__ b_i, const float* __restrict__ b_f,
    const float* __restrict__ b_a, const float* __restrict__ b_o,
    float* __restrict__ X)           // [32768][4096] f32
{
  __shared__ __bf16 Asm[2][128][32];
  __shared__ __bf16 Bsm[2][128][32];

  const int tid   = threadIdx.x;
  const int wave  = tid >> 5, lane = tid & 31;
  const int waveM = wave >> 2, waveN = wave & 3;
  const int ln    = lane & 15, hi16 = lane >> 4;
  const int Mbase = (blockIdx.x >> 5) * 128;
  const int Nbase = (blockIdx.x & 31) * 128;
  const int sm  = tid >> 1;
  const int sko = (tid & 1) * 16;

  v8f acc[4][2] = {};

  auto issue = [&](int buf, int kc) {
    const __bf16* ga = Ab + (size_t)(Mbase + sm) * KDIM + kc + sko;
    async_load_b128(ga,     &Asm[buf][sm][sko]);
    async_load_b128(ga + 8, &Asm[buf][sm][sko + 8]);
    const __bf16* gb = Bw + (size_t)(Nbase + sm) * KDIM + kc + sko;
    async_load_b128(gb,     &Bsm[buf][sm][sko]);
    async_load_b128(gb + 8, &Bsm[buf][sm][sko + 8]);
    if (kc + 64 < KDIM) {
      __builtin_prefetch(ga + 64, 0, 0);
      __builtin_prefetch(gb + 64, 0, 0);
    }
  };
  auto compute = [&](int buf) {
#pragma unroll
    for (int ns = 0; ns < 2; ++ns) {
      const int ncol = waveN * 32 + ns * 16 + ln;
      v16bf bfrag = concat8(*(const v8bf*)&Bsm[buf][ncol][hi16 * 8],
                            *(const v8bf*)&Bsm[buf][ncol][hi16 * 8 + 16]);
#pragma unroll
      for (int ms = 0; ms < 4; ++ms) {
        const int mrow = waveM * 64 + ms * 16 + ln;
        v16bf afrag = concat8(*(const v8bf*)&Asm[buf][mrow][hi16 * 8],
                              *(const v8bf*)&Asm[buf][mrow][hi16 * 8 + 16]);
        acc[ms][ns] = __builtin_amdgcn_wmma_f32_16x16x32_bf16(
            false, afrag, false, bfrag, (short)0, acc[ms][ns], false, false);
      }
    }
  };

  issue(0, 0);
  wait_async0();
  __syncthreads();
  const int NK = KDIM / 32;
#pragma unroll 1
  for (int kci = 0; kci < NK; ++kci) {
    const int cur = kci & 1;
    if (kci + 1 < NK) issue(cur ^ 1, (kci + 1) * 32);
    compute(cur);
    wait_async0();
    __syncthreads();
  }

#pragma unroll
  for (int ns = 0; ns < 2; ++ns) {
    const int n  = Nbase + waveN * 32 + ns * 16 + ln;
    const int g  = n >> 10, hc = n & 1023;
    const float bias = (g == 0) ? b_i[hc] : (g == 1) ? b_f[hc]
                      : (g == 2) ? b_a[hc] : b_o[hc];
#pragma unroll
    for (int ms = 0; ms < 4; ++ms) {
#pragma unroll
      for (int v = 0; v < 8; ++v) {
        const int row = Mbase + waveM * 64 + ms * 16 + v + hi16 * 8;
        X[(size_t)row * FOURH + n] = acc[ms][ns][v] + bias;
      }
    }
  }
}

// --------------------------- K4: persistent recurrence ---------------------
// Grid = 64 WGs; WG owns 16 hidden cols x 4 gates (64 Wh rows, LDS-resident).
// 8 waves: wave w -> gate g = w>>1, M-half = w&1 (2 Msub of 16 rows each).
// Dynamic LDS: WhL 128KB | hL 128KB | gbuf 16KB  (272KB of 320KB).
__global__ __launch_bounds__(256) void lstm_persistent_kernel(
    __bf16* __restrict__ hb0,          // [64][1024] bf16 ping
    __bf16* __restrict__ hb1,          // [64][1024] bf16 pong
    const __bf16* __restrict__ Wh,     // [4096][1024] bf16 (n-major)
    const float*  __restrict__ X,      // [T*B][4096] f32
    const float*  __restrict__ cx,     // [64][1024] f32 initial c
    float* __restrict__ out,           // outputs | hy | cy
    unsigned* __restrict__ bar)
{
  extern __shared__ char smem[];
  __bf16 (*WhL)[KDIM] = (__bf16(*)[KDIM])(smem);            // [64][1024]
  __bf16 (*hL)[KDIM]  = (__bf16(*)[KDIM])(smem + 131072);   // [64][1024]
  float  (*gbuf)[64][16] = (float(*)[64][16])(smem + 262144); // [4][64][16]

  const int tid   = threadIdx.x;
  const int wave  = tid >> 5, lane = tid & 31;
  const int ln    = lane & 15, hi16 = lane >> 4;
  const int g     = wave >> 1;
  const int mhalf = wave & 1;
  const int hbase = blockIdx.x * 16;

  // ---- one-time: pin this WG's Wh slice (64 rows x 1024 k) in LDS ----
  {
    const int r    = tid >> 2;            // 0..63 local Wh row
    const int kbb  = (tid & 3) * 256;     // 256-k span per thread
    const int grow = (r >> 4) * 1024 + hbase + (r & 15);
    const __bf16* src = Wh + (size_t)grow * KDIM + kbb;
#pragma unroll
    for (int q = 0; q < 32; ++q)
      async_load_b128(src + q * 8, &WhL[r][kbb + q * 8]);
  }
  wait_async0();
  __syncthreads();

  // ---- persistent c state in registers (4 cells per thread) ----
  const int em  = tid >> 2;               // elementwise row 0..63
  const int ej0 = (tid & 3) * 4;          // 4 cols per thread
  float creg[4];
#pragma unroll
  for (int j = 0; j < 4; ++j)
    creg[j] = cx[(size_t)em * HID + hbase + ej0 + j];

  const size_t HY = (size_t)T_STEPS * BATCH * HID;

  // h-chunk loader: 64 rows x 256 k per chunk, 8 async b128 per thread
  const int hr  = tid >> 2;
  const int hks = (tid & 3) * 64;

#pragma unroll 1
  for (int t = 0; t < T_STEPS; ++t) {
    const __bf16* hin  = (t & 1) ? hb1 : hb0;
    __bf16*       hout = (t & 1) ? hb0 : hb1;
    const float*  Xt   = X + (size_t)t * BATCH * FOURH;

    // accumulators seeded with x-part pregates
    v8f acc[2];
#pragma unroll
    for (int ms = 0; ms < 2; ++ms)
#pragma unroll
      for (int v = 0; v < 8; ++v)
        acc[ms][v] = Xt[(size_t)((mhalf * 2 + ms) * 16 + v + hi16 * 8) * FOURH
                        + g * 1024 + hbase + ln];

    auto issue_h = [&](int c) {
      const int kb = c * 256 + hks;
      const __bf16* src = hin + (size_t)hr * KDIM + kb;
#pragma unroll
      for (int q = 0; q < 8; ++q)
        async_load_b128(src + q * 8, &hL[hr][kb + q * 8]);
    };

    issue_h(0);
#pragma unroll 1
    for (int c = 0; c < 4; ++c) {
      if (c < 3) { issue_h(c + 1); wait_async8(); }  // keep next in flight
      else       { wait_async0(); }
      __syncthreads();
#pragma unroll
      for (int kc = c * 256; kc < (c + 1) * 256; kc += 32) {
        v16bf bfrag = concat8(*(const v8bf*)&WhL[g * 16 + ln][kc + hi16 * 8],
                              *(const v8bf*)&WhL[g * 16 + ln][kc + hi16 * 8 + 16]);
#pragma unroll
        for (int ms = 0; ms < 2; ++ms) {
          const int mrow = (mhalf * 2 + ms) * 16 + ln;
          v16bf afrag = concat8(*(const v8bf*)&hL[mrow][kc + hi16 * 8],
                                *(const v8bf*)&hL[mrow][kc + hi16 * 8 + 16]);
          acc[ms] = __builtin_amdgcn_wmma_f32_16x16x32_bf16(
              false, afrag, false, bfrag, (short)0, acc[ms], false, false);
        }
      }
    }

    // gate exchange through LDS
#pragma unroll
    for (int ms = 0; ms < 2; ++ms)
#pragma unroll
      for (int v = 0; v < 8; ++v)
        gbuf[g][(mhalf * 2 + ms) * 16 + v + hi16 * 8][ln] = acc[ms][v];
    __syncthreads();

    // elementwise: c in registers, h to global (out + bf16 ping-pong)
#pragma unroll
    for (int j = 0; j < 4; ++j) {
      const int hc  = ej0 + j;
      const int col = hbase + hc;
      const float ig = sigmoidf_(gbuf[0][em][hc]);
      const float fg = sigmoidf_(gbuf[1][em][hc]);
      const float ag = tanhf(gbuf[2][em][hc]);
      const float og = sigmoidf_(gbuf[3][em][hc]);
      const float c  = fg * creg[j] + ig * ag;
      creg[j] = c;
      const float hv = og * tanhf(c);
      out[((size_t)t * BATCH + em) * HID + col] = hv;
      hout[(size_t)em * HID + col] = (__bf16)hv;
      if (t == T_STEPS - 1) out[HY + (size_t)em * HID + col] = hv;  // hy
    }

    // device-wide step barrier (monotonic counter, L2 atomics)
    __syncthreads();
    if (tid == 0) {
      __threadfence();
      atomicAdd(bar, 1u);
      const unsigned target = (unsigned)NWG * (unsigned)(t + 1);
      while (atomicAdd(bar, 0u) < target) __builtin_amdgcn_s_sleep(1);
      __threadfence();
    }
    __syncthreads();
  }

  // cy
#pragma unroll
  for (int j = 0; j < 4; ++j)
    out[HY + (size_t)BATCH * HID + (size_t)em * HID + hbase + ej0 + j] = creg[j];
}

// --------------------------- host launcher ---------------------------------
extern "C" void kernel_launch(void* const* d_in, const int* in_sizes, int n_in,
                              void* d_out, int out_size, void* d_ws,
                              size_t ws_size, hipStream_t stream) {
  const float* inputs = (const float*)d_in[0];
  const float* hx     = (const float*)d_in[1];
  const float* cx     = (const float*)d_in[2];
  const float* w_ii = (const float*)d_in[3];
  const float* w_fi = (const float*)d_in[4];
  const float* w_ai = (const float*)d_in[5];
  const float* w_oi = (const float*)d_in[6];
  const float* w_ih = (const float*)d_in[7];
  const float* w_fh = (const float*)d_in[8];
  const float* w_ah = (const float*)d_in[9];
  const float* w_oh = (const float*)d_in[10];
  const float* b_i  = (const float*)d_in[11];
  const float* b_f  = (const float*)d_in[12];
  const float* b_a  = (const float*)d_in[13];
  const float* b_o  = (const float*)d_in[14];
  float* out = (float*)d_out;

  // workspace layout
  char*     ws   = (char*)d_ws;
  __bf16*   Wi   = (__bf16*)ws;                           //   8 MB
  __bf16*   Wh   = (__bf16*)(ws + (8ull << 20));          //   8 MB
  __bf16*   hb0  = (__bf16*)(ws + (16ull << 20));         // 128 KB
  __bf16*   hb1  = hb0 + (size_t)BATCH * HID;             // 128 KB
  unsigned* bar  = (unsigned*)(ws + (16ull << 20) + 262144);
  __bf16*   Ab   = (__bf16*)(ws + (24ull << 20));         //  64 MB
  float*    X    = (float*)(ws + (96ull << 20));          // 512 MB

  pack_weights_kernel<<<8192, 256, 0, stream>>>(
      w_ii, w_fi, w_ai, w_oi, w_ih, w_fh, w_ah, w_oh, Wi, Wh);
  pack_inputs_kernel<<<32768, 256, 0, stream>>>(inputs, Ab);
  init_state_kernel<<<256, 256, 0, stream>>>(hx, hb0, bar);
  gemm_x_kernel<<<8192, 256, 0, stream>>>(Ab, Wi, b_i, b_f, b_a, b_o, X);

  lstm_persistent_kernel<<<NWG, 256, 272 * 1024, stream>>>(
      hb0, hb1, Wh, X, cx, out, bar);
}
